// DST_1133871366830
// MI455X (gfx1250) — matrix operand order repaired and checked by reading, not compiled, not run on hardware
//
#include <hip/hip_runtime.h>
#include <math.h>

// DST-II as GEMM: Y = X @ S,  S[n,k] = sin(pi/N * (n+0.5) * (k+1))
// X: (16384, 1024) f32.  Split-f16 WMMA (hi/lo) for ~f32 accuracy:
//   x*S ~= xh*Sh + xh*Sl + xl*Sh   (3x v_wmma_f32_16x16x32_f16, f32 accum)

typedef __attribute__((ext_vector_type(16))) _Float16 v16h;
typedef __attribute__((ext_vector_type(8)))  float    v8f;

#define M_DIM 16384
#define N_DIM 1024
#define KB    32     // K step
#define BM    128    // block M tile
#define BN    128    // block N tile
#define LDA   40     // padded LDS row stride (halves) -> 80B rows, kills bank conflicts

union FragU { v16h v; uint4 q[2]; };

// ---------------------------------------------------------------------------
// Kernel 1: generate sine basis, pre-packed in WMMA B-fragment layout.
// Fragment (kt, nt): lane l (h = l/16) holds S[kt*32 + k(h,i)][nt*16 + l%16],
// k(h,i) = 8h+i (i<8), 16+8h+i-8 (i>=8), matching the A fragment K order.
// hi = f16(s), lo = f16(s - (f32)hi).
// ---------------------------------------------------------------------------
__global__ __launch_bounds__(256) void dst_gen_basis(_Float16* __restrict__ bhi,
                                                     _Float16* __restrict__ blo) {
    int t    = blockIdx.x * blockDim.x + threadIdx.x; // 0..65535
    int lane = t & 31;
    int frag = t >> 5;                                // kt*64 + nt
    int nt   = frag & 63;
    int kt   = frag >> 6;
    int lh   = lane >> 4;

    FragU h, l;
#pragma unroll
    for (int i = 0; i < 16; ++i) {
        float a    = (float)(kt * 32 + (i < 8 ? 8 * lh + i : 16 + 8 * lh + (i - 8))) + 0.5f; // n + 0.5 (exact)
        float kk   = (float)(nt * 16 + (lane & 15) + 1); // k + 1 (exact)
        float prod = a * kk;                          // <= ~2^20: exact in f32
        float ph   = fmodf(prod, 2048.0f);            // reduce by period 2N (exact)
        float s    = sinf(ph * 3.0679615757712823e-3f); // * pi/1024
        _Float16 sh = (_Float16)s;
        _Float16 sl = (_Float16)(s - (float)sh);
        h.v[i] = sh;
        l.v[i] = sl;
    }
    size_t base = ((size_t)frag * 32 + lane) * 16;
    *(uint4*)(bhi + base)     = h.q[0];
    *(uint4*)(bhi + base + 8) = h.q[1];
    *(uint4*)(blo + base)     = l.q[0];
    *(uint4*)(blo + base + 8) = l.q[1];
}

// ---------------------------------------------------------------------------
// Kernel 2: blocked WMMA GEMM. 256 threads = 8 wave32. Block = 128x128 out.
// Wave grid 2(M) x 4(N): wave tile = 64(M) x 32(N) = 4x2 accumulators of
// 16x16 (64 VGPRs). Balances DS (A-frag) vs vector-L2 (B-frag) traffic:
// B redundancy 2x (was 4x), A served from LDS which is built for sharing.
// ---------------------------------------------------------------------------
__global__ __launch_bounds__(256) void dst_gemm(const float*    __restrict__ x,
                                                const _Float16* __restrict__ bhi,
                                                const _Float16* __restrict__ blo,
                                                float*          __restrict__ y) {
    __shared__ alignas(16) _Float16 sAhi[BM * LDA];
    __shared__ alignas(16) _Float16 sAlo[BM * LDA];

    const int t     = threadIdx.x;
    const int lane  = t & 31;
    const int w     = t >> 5;       // wave 0..7
    const int wm    = w & 1;        // M sub-block (64 rows each)
    const int wn    = w >> 1;       // N sub-block (32 cols each)
    const int hlane = lane & 15;
    const int chunk = lane >> 4;    // A-frag K-chunk select (ISA 16-bit A layout)

    const int rowBlk = blockIdx.y * BM;
    const int ntBase = blockIdx.x * 8 + wn * 2;  // base 16-wide N tile index

    // cooperative staging: 2 threads per row, 16 f32 each (128 x 32 f32 tile)
    const int srow = t >> 1;
    const int skh  = (t & 1) * 16;
    const float* gx = x + (size_t)(rowBlk + srow) * N_DIM + skh;

    v8f acc[4][2] = {};

    for (int kb = 0; kb < N_DIM / KB; ++kb) {
        // ---- stage X tile: f32 -> (hi,lo) f16 in LDS -------------------
        const float4* g4 = (const float4*)(gx + kb * KB);
        float4 f0 = g4[0], f1 = g4[1], f2 = g4[2], f3 = g4[3];
        float fv[16] = {f0.x, f0.y, f0.z, f0.w, f1.x, f1.y, f1.z, f1.w,
                        f2.x, f2.y, f2.z, f2.w, f3.x, f3.y, f3.z, f3.w};
        _Float16* dh = &sAhi[srow * LDA + skh];
        _Float16* dl = &sAlo[srow * LDA + skh];
#pragma unroll
        for (int i = 0; i < 16; ++i) {
            _Float16 hh = (_Float16)fv[i];
            dh[i] = hh;
            dl[i] = (_Float16)(fv[i] - (float)hh);
        }
        __syncthreads();

        // ---- A fragments from LDS (two ds_load_b128 per frag) ----------
        // lanes 0-15: K chunks {0,2}; lanes 16-31: {1,3}; row = M
        v16h ahi[4], alo[4];
#pragma unroll
        for (int mt = 0; mt < 4; ++mt) {
            int r = wm * 64 + mt * 16 + hlane;
            const uint4* ph = (const uint4*)&sAhi[r * LDA + chunk * 8];
            const uint4* pl = (const uint4*)&sAlo[r * LDA + chunk * 8];
            FragU fa; fa.q[0] = ph[0]; fa.q[1] = ph[2]; ahi[mt] = fa.v; // +32B = chunk+2
            FragU fb; fb.q[0] = pl[0]; fb.q[1] = pl[2]; alo[mt] = fb.v;
        }

        // ---- B fragments (prepacked, L2-resident) + WMMA ---------------
#pragma unroll
        for (int j = 0; j < 2; ++j) {
            size_t fi = (((size_t)kb * 64) + ntBase + j) * 32 + lane;
            const uint4* pb = (const uint4*)(bhi + fi * 16);
            const uint4* pc = (const uint4*)(blo + fi * 16);
            FragU fh; fh.q[0] = pb[0]; fh.q[1] = pb[1];
            FragU fl; fl.q[0] = pc[0]; fl.q[1] = pc[1];
#pragma unroll
            for (int mt = 0; mt < 4; ++mt) {
                acc[mt][j] = __builtin_amdgcn_wmma_f32_16x16x32_f16(
                    false, ahi[mt], false, fh.v, (short)0, acc[mt][j], false, false);
                acc[mt][j] = __builtin_amdgcn_wmma_f32_16x16x32_f16(
                    false, ahi[mt], false, fl.v, (short)0, acc[mt][j], false, false);
                acc[mt][j] = __builtin_amdgcn_wmma_f32_16x16x32_f16(
                    false, alo[mt], false, fh.v, (short)0, acc[mt][j], false, false);
            }
        }

        // prefetch next K-step's B region (WGP-scope: into near cache too)
        if (kb < N_DIM / KB - 1) {
            size_t nf = (((size_t)(kb + 1) * 64) + ntBase) * 32 + lane;
            __builtin_prefetch(bhi + nf * 16, 0, 3);
            __builtin_prefetch(blo + nf * 16, 0, 3);
        }
        __syncthreads();
    }

    // ---- epilogue: C/D 16x16 f32 layout: VGPR r -> M = r (lanes<16) / r+8 ---
#pragma unroll
    for (int mt = 0; mt < 4; ++mt) {
        int row0 = rowBlk + wm * 64 + mt * 16 + chunk * 8;
#pragma unroll
        for (int j = 0; j < 2; ++j) {
            int col = blockIdx.x * BN + wn * 32 + j * 16 + hlane;
            float* yo = y + (size_t)row0 * N_DIM + col;
            v8f c = acc[mt][j];
#pragma unroll
            for (int r = 0; r < 8; ++r)
                yo[(size_t)r * N_DIM] = c[r];
        }
    }
}

// ---------------------------------------------------------------------------
extern "C" void kernel_launch(void* const* d_in, const int* in_sizes, int n_in,
                              void* d_out, int out_size, void* d_ws, size_t ws_size,
                              hipStream_t stream) {
    (void)in_sizes; (void)n_in; (void)out_size; (void)ws_size;
    const float* x = (const float*)d_in[0];
    float*       y = (float*)d_out;

    // workspace: packed sine basis, hi + lo f16 (2 MB each)
    _Float16* bhi = (_Float16*)d_ws;
    _Float16* blo = bhi + (size_t)N_DIM * N_DIM;

    // 1M basis elements: 65536 threads x 16 sin each
    dst_gen_basis<<<256, 256, 0, stream>>>(bhi, blo);

    dim3 grid(N_DIM / BN, M_DIM / BM);   // x = N blocks (inner sweep -> L2 reuse of X)
    dst_gemm<<<grid, 256, 0, stream>>>(x, bhi, blo, y);
}
